// InternLM3Attention_10634339025329
// MI455X (gfx1250) — matrix operand order repaired and checked by reading, not compiled, not run on hardware
//
// InternLM3 attention layer for MI455X (gfx1250, CDNA5, wave32).
//
// Pipeline (all bf16-input / f32-accumulate WMMA, v_wmma_f32_16x16x32_bf16):
//   1) convert X, Wq, Wk, Wv, Wo  f32 -> bf16            (bandwidth pass, ~15us)
//   2) GEMM  Q = X@Wq, K = X@Wk, V = X@Wv   (f32 out)    (TDM-staged WMMA GEMM)
//   3) RoPE + transpose to [B,H,S,HD] bf16 for Q,K; transpose+convert V
//   4) flash attention (online softmax, causal), WMMA for QK^T and PV,
//      V tiles double-buffered through LDS via the Tensor Data Mover
//   5) GEMM  out = attn@Wo  (f32 out)
//
// ~481 GFLOP vs ~0.4GB traffic -> compute bound -> keep the bf16 WMMA pipe
// busy and hide all global->LDS staging behind TDM (TENSORcnt) double
// buffering.  LDS tiles use TDM pad (1 dword per 16/64-dword interval) so
// column-wise fragment reads are bank-conflict free.

#include <hip/hip_runtime.h>

#define BB      2
#define SS      2048
#define HIDDEN  4096
#define NHEADS  32
#define NKVH    8
#define HDIM    128
#define QDIM    (NHEADS * HDIM)   // 4096
#define KVDIM   (NKVH * HDIM)     // 1024
#define TOKENS  (BB * SS)         // 4096
#define ATT_SCALE 0.08838834764831845f
#define NEG_BIG  (-1.0e30f)

typedef __attribute__((ext_vector_type(16))) __bf16 bf16x16;
typedef __attribute__((ext_vector_type(8)))  float  floatx8;
typedef __attribute__((ext_vector_type(4)))  unsigned int uint32x4;
typedef __attribute__((ext_vector_type(8)))  int   int32x8;
typedef __attribute__((ext_vector_type(4)))  int   int32x4;

#if defined(__has_builtin)
#if __has_builtin(__builtin_amdgcn_tensor_load_to_lds) && \
    __has_builtin(__builtin_amdgcn_s_wait_tensorcnt)
#define HAVE_TDM 1
#endif
#endif
#ifndef HAVE_TDM
#define HAVE_TDM 0
#endif

__device__ __forceinline__ floatx8 zero8() {
  floatx8 z;
#pragma unroll
  for (int i = 0; i < 8; ++i) z[i] = 0.0f;
  return z;
}

// A-fragment (16x32 bf16): lane r=lane&15 holds row r; lane-half kh picks
// K range [16*kh .. 16*kh+15] (same k order as the B fragments).
__device__ __forceinline__ bf16x16 load_a16(const __bf16* row, int kh) {
  bf16x16 a;
#pragma unroll
  for (int e = 0; e < 16; ++e) a[e] = row[kh * 16 + e];
  return a;
}

// B-fragment (32x16 bf16) from 16 contiguous K values (lane n=lane&15 is the
// column, lane-half kh picks K range [16*kh .. 16*kh+15]).
__device__ __forceinline__ bf16x16 load_b16(const __bf16* p) {
  bf16x16 b;
#pragma unroll
  for (int e = 0; e < 16; ++e) b[e] = p[e];
  return b;
}

// B-fragment from an LDS tile in natural [k][n] layout with `pitch` bf16 per
// row (pitch = 130 -> 65-dword stride -> bank-conflict-free column reads).
__device__ __forceinline__ bf16x16 load_b16_strided(const __bf16* base, int pitch,
                                                    int n, int kh) {
  bf16x16 b;
#pragma unroll
  for (int e = 0; e < 16; ++e) b[e] = base[(16 * kh + e) * pitch + n];
  return b;
}

__device__ __forceinline__ floatx8 wmma_bf16(bf16x16 a, bf16x16 b, floatx8 c) {
  return __builtin_amdgcn_wmma_f32_16x16x32_bf16(false, a, false, b,
                                                 (short)0, c, false, false);
}

#if HAVE_TDM
// ---------------------------------------------------------------------------
// Tensor Data Mover: async 2-D tile load global -> LDS (ISA ch.8 D# layout).
//   data_size = 2 bytes (bf16); dims/strides in elements; pad_enable inserts
//   1 dword every 2^(pad_code+1) dwords so the LDS tile gets a conflict-free
//   pitch.  Groups 2/3 zeroed (2-D tensor, no iteration / gather).
// ---------------------------------------------------------------------------
__device__ __forceinline__ void tdm_load_2d(unsigned lds_addr, const void* gptr,
                                            unsigned tensor_d0, unsigned tensor_d1,
                                            unsigned d0_stride,
                                            unsigned tile_d0, unsigned tile_d1,
                                            unsigned pad_code) {
  unsigned long long ga = (unsigned long long)(size_t)gptr;
  uint32x4 g0;
  g0[0] = 1u;                                         // count=1 (valid, user)
  g0[1] = lds_addr;                                   // LDS byte address
  g0[2] = (unsigned)(ga & 0xFFFFFFFFu);               // global_addr[31:0]
  g0[3] = (unsigned)((ga >> 32) & 0x01FFFFFFu) | (2u << 30);  // addr hi | type=2
  int32x8 g1;
  g1[0] = (int)((1u << 16) |                          // data_size = 1 -> 2 bytes
                (1u << 20) |                          // pad_enable
                (pad_code << 22));                    // pad_interval; amount=0 (1 dw)
  g1[1] = (int)((tensor_d0 & 0xFFFFu) << 16);         // dim0[15:0]
  g1[2] = (int)(((tensor_d0 >> 16) & 0xFFFFu) | ((tensor_d1 & 0xFFFFu) << 16));
  g1[3] = (int)(((tensor_d1 >> 16) & 0xFFFFu) | ((tile_d0 & 0xFFFFu) << 16));
  g1[4] = (int)(tile_d1 & 0xFFFFu);                   // tile_dim1; tile_dim2=0
  g1[5] = (int)d0_stride;                             // dim0_stride[31:0]
  g1[6] = 0;                                          // stride hi, dim1_stride lo
  g1[7] = 0;
  int32x4 gz = {0, 0, 0, 0};
#if __clang_major__ >= 23
  int32x8 gz8 = {0, 0, 0, 0, 0, 0, 0, 0};
  __builtin_amdgcn_tensor_load_to_lds(g0, g1, gz, gz, gz8, 0);
#else
  __builtin_amdgcn_tensor_load_to_lds(g0, g1, gz, gz, 0);
#endif
}

__device__ __forceinline__ unsigned lds_off(const void* p) {
  return (unsigned)(size_t)p;  // low 32 bits of a generic LDS addr = LDS offset
}
#endif  // HAVE_TDM

// ---------------------------------------------------------------------------
// f32 -> bf16 convert
// ---------------------------------------------------------------------------
__global__ void f32_to_bf16_kernel(const float* __restrict__ in,
                                   __bf16* __restrict__ out, size_t n) {
  size_t i = (size_t)blockIdx.x * blockDim.x + threadIdx.x;
  if (i < n) out[i] = (__bf16)in[i];
}

// ---------------------------------------------------------------------------
// bf16 WMMA GEMM:  C[M,N] f32 = A[M,K] bf16 @ B[K,N] bf16 (both row major).
// Block = 256 threads = 8 waves (4Mx2N); tile 128x128x32; wave = 32x64 =
// 2x4 WMMA tiles.  A tile [m][k] pitch 34, B tile [k][n] pitch 130.
// TDM path: double-buffered async staging hidden under WMMA.
// ---------------------------------------------------------------------------
__global__ __launch_bounds__(256) void gemm_bf16_kernel(
    const __bf16* __restrict__ A, const __bf16* __restrict__ B,
    float* __restrict__ C, int M, int N, int K) {
#if HAVE_TDM
  __shared__ __attribute__((aligned(16))) __bf16 sA[2][128 * 34];
  __shared__ __attribute__((aligned(16))) __bf16 sB[2][32 * 130];
#else
  __shared__ __attribute__((aligned(16))) __bf16 sA1[128 * 34];
  __shared__ __attribute__((aligned(16))) __bf16 sB1[32 * 130];
#endif

  const int tid  = threadIdx.x;
  const int lane = tid & 31;
  const int wave = tid >> 5;
  const int wm = wave >> 1, wn = wave & 1;
  const int r = lane & 15, kh = lane >> 4;
  const int row0 = blockIdx.y * 128;
  const int col0 = blockIdx.x * 128;

  floatx8 acc[2][4];
#pragma unroll
  for (int mi = 0; mi < 2; ++mi)
#pragma unroll
    for (int ni = 0; ni < 4; ++ni) acc[mi][ni] = zero8();

#if HAVE_TDM
  if (wave == 0) {  // prologue: fill buffer 0, synchronously
    tdm_load_2d(lds_off(&sA[0][0]), A + (size_t)row0 * K, K, M, K, 32, 128, 3);
    tdm_load_2d(lds_off(&sB[0][0]), B + col0, N, K, N, 128, 32, 5);
    __builtin_amdgcn_s_wait_tensorcnt(0);
  }
  __syncthreads();
#endif

  for (int k0 = 0, kt = 0; k0 < K; k0 += 32, ++kt) {
#if HAVE_TDM
    const __bf16* a_s = sA[kt & 1];
    const __bf16* b_s = sB[kt & 1];
    if (wave == 0 && k0 + 32 < K) {  // async-prefetch next K tile (other buf)
      tdm_load_2d(lds_off(&sA[(kt + 1) & 1][0]),
                  A + (size_t)row0 * K + k0 + 32, K, M, K, 32, 128, 3);
      tdm_load_2d(lds_off(&sB[(kt + 1) & 1][0]),
                  B + (size_t)(k0 + 32) * N + col0, N, K, N, 128, 32, 5);
    }
#else
    // Manual staging fallback (dword copies, same LDS layouts).
    {
      const int arow = tid >> 1, half = tid & 1;
      const unsigned int* g =
          (const unsigned int*)(A + (size_t)(row0 + arow) * K + k0 + half * 16);
      unsigned int* s = (unsigned int*)(sA1 + arow * 34 + half * 16);
#pragma unroll
      for (int i = 0; i < 8; ++i) s[i] = g[i];
    }
    {
      const int kk = tid >> 3, n0 = (tid & 7) * 16;
      const unsigned int* g =
          (const unsigned int*)(B + (size_t)(k0 + kk) * N + col0 + n0);
      unsigned int* s = (unsigned int*)(sB1 + kk * 130 + n0);
#pragma unroll
      for (int i = 0; i < 8; ++i) s[i] = g[i];
    }
    if (k0 + 32 < K) {
      __builtin_prefetch((const void*)(A + (size_t)(row0 + (tid >> 1)) * K + k0 + 32), 0, 1);
      __builtin_prefetch((const void*)(B + (size_t)(k0 + 32 + (tid >> 3)) * N + col0 + (tid & 7) * 16), 0, 1);
    }
    __syncthreads();
    const __bf16* a_s = sA1;
    const __bf16* b_s = sB1;
#endif

    bf16x16 af[2], bfrag[4];
#pragma unroll
    for (int mi = 0; mi < 2; ++mi)
      af[mi] = load_a16(a_s + (wm * 32 + mi * 16 + r) * 34, kh);
#pragma unroll
    for (int ni = 0; ni < 4; ++ni)
      bfrag[ni] = load_b16_strided(b_s, 130, wn * 64 + ni * 16 + r, kh);

#pragma unroll
    for (int mi = 0; mi < 2; ++mi)
#pragma unroll
      for (int ni = 0; ni < 4; ++ni)
        acc[mi][ni] = wmma_bf16(af[mi], bfrag[ni], acc[mi][ni]);

#if HAVE_TDM
    if (wave == 0) __builtin_amdgcn_s_wait_tensorcnt(0);
#endif
    __syncthreads();
  }

  // Epilogue: C/D layout -> lane holds col n=r, rows m = i + 8*kh.
#pragma unroll
  for (int mi = 0; mi < 2; ++mi)
#pragma unroll
    for (int ni = 0; ni < 4; ++ni)
#pragma unroll
      for (int i = 0; i < 8; ++i) {
        int m = row0 + wm * 32 + mi * 16 + i + 8 * kh;
        int n = col0 + wn * 64 + ni * 16 + r;
        C[(size_t)m * N + n] = acc[mi][ni][i];
      }
}

// ---------------------------------------------------------------------------
// RoPE + transpose to head-major bf16:  in f32 [B,S,nh,HD] -> out bf16
// [B,nh,S,HD].  rotate_half: d<64 -> -x[d+64], d>=64 -> x[d-64].
// ---------------------------------------------------------------------------
__global__ void rope_transpose_kernel(const float* __restrict__ x,
                                      const int* __restrict__ pos,
                                      const float* __restrict__ cosT,
                                      const float* __restrict__ sinT,
                                      __bf16* __restrict__ out, int nh) {
  size_t idx = (size_t)blockIdx.x * blockDim.x + threadIdx.x;
  size_t total = (size_t)BB * SS * nh * HDIM;
  if (idx >= total) return;
  int d = idx & (HDIM - 1);
  int h = (idx >> 7) % nh;
  int s = (idx / ((size_t)HDIM * nh)) % SS;
  int b = idx / ((size_t)HDIM * nh * SS);
  int p = pos[b * SS + s];
  float c  = cosT[(size_t)p * HDIM + d];
  float sn = sinT[(size_t)p * HDIM + d];
  size_t base = ((size_t)(b * SS + s) * nh + h) * HDIM;
  float xv = x[base + d];
  float ot = (d < HDIM / 2) ? -x[base + d + HDIM / 2] : x[base + d - HDIM / 2];
  out[(((size_t)(b * nh + h)) * SS + s) * HDIM + d] = (__bf16)(xv * c + ot * sn);
}

__global__ void transpose_bf16_kernel(const float* __restrict__ x,
                                      __bf16* __restrict__ out, int nh) {
  size_t idx = (size_t)blockIdx.x * blockDim.x + threadIdx.x;
  size_t total = (size_t)BB * SS * nh * HDIM;
  if (idx >= total) return;
  int d = idx & (HDIM - 1);
  int h = (idx >> 7) % nh;
  int s = (idx / ((size_t)HDIM * nh)) % SS;
  int b = idx / ((size_t)HDIM * nh * SS);
  out[(((size_t)(b * nh + h)) * SS + s) * HDIM + d] =
      (__bf16)x[((size_t)(b * SS + s) * nh + h) * HDIM + d];
}

// ---------------------------------------------------------------------------
// Flash attention, causal, GQA.  One wave per (b, head, 16-row Q tile).
// KV processed in 32-wide tiles: 8 WMMAs for QK^T (K=128) + 8 WMMAs for PV
// (HD=128).  Online softmax in f32 registers.  V tiles double-buffered via
// TDM (issue next tile, compute QK^T, s_wait_tensorcnt 1, consume current).
// ---------------------------------------------------------------------------
__global__ __launch_bounds__(32) void flash_attn_kernel(
    const __bf16* __restrict__ Q,   // [B,NH,S,HD]
    const __bf16* __restrict__ Kk,  // [B,NKV,S,HD]
    const __bf16* __restrict__ V,   // [B,NKV,S,HD]
    __bf16* __restrict__ attn) {    // [B*S, NH*HD]
  const int lane = threadIdx.x & 31;
  const int r = lane & 15, kh = lane >> 4;
  const int qtiles = SS / 16;
  int bid = blockIdx.x;
  int qt = bid % qtiles;
  int h  = (bid / qtiles) % NHEADS;
  int b  = bid / (qtiles * NHEADS);
  int hk = h / (NHEADS / NKVH);

  const __bf16* Qh = Q  + (size_t)(b * NHEADS + h) * SS * HDIM;
  const __bf16* Kh = Kk + (size_t)(b * NKVH + hk) * SS * HDIM;
  const __bf16* Vh = V  + (size_t)(b * NKVH + hk) * SS * HDIM;

  __shared__ __attribute__((aligned(16))) __bf16 p_lds[16 * 34];
#if HAVE_TDM
  __shared__ __attribute__((aligned(16))) __bf16 v_lds[2][32 * 130];
#else
  __shared__ __attribute__((aligned(16))) __bf16 v_lds1[32 * 130];
#endif

  // Preload Q A-fragments (16 rows x 128 K, as 4 chunks of 32).
  bf16x16 qa[4];
  const int qrow = qt * 16 + r;
#pragma unroll
  for (int c = 0; c < 4; ++c)
    qa[c] = load_a16(Qh + (size_t)qrow * HDIM + c * 32, kh);

  floatx8 o[8];
#pragma unroll
  for (int ni = 0; ni < 8; ++ni) o[ni] = zero8();
  float mrow[8], lrow[8];
#pragma unroll
  for (int i = 0; i < 8; ++i) { mrow[i] = NEG_BIG; lrow[i] = 0.0f; }

  const int ntiles = (qt + 2) >> 1;  // ceil((qt*16+16)/32)

#if HAVE_TDM
  // V tile: 32 rows x 128 bf16, pitch 130 (65-dword rows via TDM pad).
  tdm_load_2d(lds_off(&v_lds[0][0]), Vh, HDIM, SS, HDIM, HDIM, 32, 5);
#endif

  for (int kt = 0; kt < ntiles; ++kt) {
    const int kv0 = kt * 32;
#if HAVE_TDM
    const __bf16* vbuf = v_lds[kt & 1];
    const bool more = (kt + 1 < ntiles);
    if (more)
      tdm_load_2d(lds_off(&v_lds[(kt + 1) & 1][0]),
                  Vh + (size_t)(kv0 + 32) * HDIM, HDIM, SS, HDIM, HDIM, 32, 5);
#else
    const __bf16* vbuf = v_lds1;
    {  // manual staging: 2048 dwords, coalesced, 65-dword pitch
      const unsigned int* gv = (const unsigned int*)(Vh + (size_t)kv0 * HDIM);
      unsigned int* sv = (unsigned int*)v_lds1;
#pragma unroll 8
      for (int j = 0; j < 64; ++j) {
        int dw = lane + 32 * j;
        sv[(dw >> 6) * 65 + (dw & 63)] = gv[dw];
      }
    }
#endif

    // Scores: two 16-col tiles over the 32-wide KV slab (overlaps V DMA).
    floatx8 s0 = zero8(), s1 = zero8();
#pragma unroll
    for (int c = 0; c < 4; ++c) {
      bf16x16 kb0 = load_b16(Kh + (size_t)(kv0 + r) * HDIM + c * 32 + kh * 16);
      bf16x16 kb1 = load_b16(Kh + (size_t)(kv0 + 16 + r) * HDIM + c * 32 + kh * 16);
      s0 = wmma_bf16(qa[c], kb0, s0);
      s1 = wmma_bf16(qa[c], kb1, s1);
    }

    const bool do_mask = (kt == ntiles - 1);  // only the diagonal tile
#pragma unroll
    for (int i = 0; i < 8; ++i) {
      const int m = qt * 16 + i + 8 * kh;     // global query row
      float x0 = s0[i] * ATT_SCALE;
      float x1 = s1[i] * ATT_SCALE;
      if (do_mask) {
        if (kv0 + r > m)      x0 = NEG_BIG;
        if (kv0 + 16 + r > m) x1 = NEG_BIG;
      }
      float mx = fmaxf(x0, x1);
      mx = fmaxf(mx, __shfl_xor(mx, 1, 32));
      mx = fmaxf(mx, __shfl_xor(mx, 2, 32));
      mx = fmaxf(mx, __shfl_xor(mx, 4, 32));
      mx = fmaxf(mx, __shfl_xor(mx, 8, 32));
      const float mnew  = fmaxf(mrow[i], mx);
      const float alpha = __expf(mrow[i] - mnew);
      const float p0 = __expf(x0 - mnew);
      const float p1 = __expf(x1 - mnew);
      float ps = p0 + p1;
      ps += __shfl_xor(ps, 1, 32);
      ps += __shfl_xor(ps, 2, 32);
      ps += __shfl_xor(ps, 4, 32);
      ps += __shfl_xor(ps, 8, 32);
      lrow[i] = lrow[i] * alpha + ps;
      mrow[i] = mnew;
#pragma unroll
      for (int ni = 0; ni < 8; ++ni) o[ni][i] *= alpha;
      // Re-stripe P into A-fragment layout via LDS.
      p_lds[(i + 8 * kh) * 34 + r]      = (__bf16)p0;
      p_lds[(i + 8 * kh) * 34 + 16 + r] = (__bf16)p1;
    }

#if HAVE_TDM
    // Current V tile done when only the just-issued DMA remains in flight.
    if (more) __builtin_amdgcn_s_wait_tensorcnt(1);
    else      __builtin_amdgcn_s_wait_tensorcnt(0);
#endif
    __syncthreads();

    // PV: A = P (16x32), B = V tile columns; accumulate 8 HD tiles.
    const bf16x16 pa = load_a16(p_lds + r * 34, kh);
#pragma unroll
    for (int ni = 0; ni < 8; ++ni) {
      bf16x16 vb = load_b16_strided(vbuf, 130, ni * 16 + r, kh);
      o[ni] = wmma_bf16(pa, vb, o[ni]);
    }
    __syncthreads();
  }

  // Normalize and write bf16 attn in token-major [B*S, NH*HD] for the O-GEMM.
#pragma unroll
  for (int ni = 0; ni < 8; ++ni)
#pragma unroll
    for (int i = 0; i < 8; ++i) {
      const int m = qt * 16 + i + 8 * kh;
      const float val = o[ni][i] / lrow[i];
      attn[(size_t)(b * SS + m) * QDIM + h * HDIM + ni * 16 + r] = (__bf16)val;
    }
}

// ---------------------------------------------------------------------------
// Host-side launcher
// ---------------------------------------------------------------------------
extern "C" void kernel_launch(void* const* d_in, const int* in_sizes, int n_in,
                              void* d_out, int out_size, void* d_ws, size_t ws_size,
                              hipStream_t stream) {
  (void)in_sizes; (void)n_in; (void)out_size; (void)ws_size;
  const float* hidden = (const float*)d_in[0];
  const int*   pos    = (const int*)d_in[1];
  const float* cosT   = (const float*)d_in[2];
  const float* sinT   = (const float*)d_in[3];
  const float* Wq     = (const float*)d_in[4];
  const float* Wk     = (const float*)d_in[5];
  const float* Wv     = (const float*)d_in[6];
  const float* Wo     = (const float*)d_in[7];
  float* out = (float*)d_out;

  char* ws = (char*)d_ws;
  size_t off = 0;
  auto take = [&](size_t bytes) -> char* {
    char* p = ws + off;
    off += (bytes + 255) & ~(size_t)255;
    return p;
  };

  __bf16* x_bf  = (__bf16*)take((size_t)TOKENS * HIDDEN * 2);  // 32MB
  __bf16* wq_bf = (__bf16*)take((size_t)HIDDEN * QDIM  * 2);   // 32MB
  __bf16* wk_bf = (__bf16*)take((size_t)HIDDEN * KVDIM * 2);   //  8MB
  __bf16* wv_bf = (__bf16*)take((size_t)HIDDEN * KVDIM * 2);   //  8MB
  __bf16* wo_bf = (__bf16*)take((size_t)QDIM  * HIDDEN * 2);   // 32MB
  float*  q_f32 = (float*) take((size_t)TOKENS * QDIM  * 4);   // 64MB (reused)
  float*  k_f32 = (float*) take((size_t)TOKENS * KVDIM * 4);   // 16MB
  float*  v_f32 = (float*) take((size_t)TOKENS * KVDIM * 4);   // 16MB
  __bf16* q_bf  = (__bf16*)take((size_t)TOKENS * QDIM  * 2);   // 32MB
  __bf16* k_bf  = (__bf16*)take((size_t)TOKENS * KVDIM * 2);   //  8MB
  __bf16* v_bf  = (__bf16*)take((size_t)TOKENS * KVDIM * 2);   //  8MB
  __bf16* attn_bf = (__bf16*)q_f32;  // q_f32 dead after RoPE; alias it.

  auto cvt = [&](const float* src, __bf16* dst, size_t n) {
    f32_to_bf16_kernel<<<(unsigned)((n + 255) / 256), 256, 0, stream>>>(src, dst, n);
  };
  cvt(hidden, x_bf,  (size_t)TOKENS * HIDDEN);
  cvt(Wq,     wq_bf, (size_t)HIDDEN * QDIM);
  cvt(Wk,     wk_bf, (size_t)HIDDEN * KVDIM);
  cvt(Wv,     wv_bf, (size_t)HIDDEN * KVDIM);
  cvt(Wo,     wo_bf, (size_t)QDIM  * HIDDEN);

  // Projections.
  gemm_bf16_kernel<<<dim3(QDIM / 128, TOKENS / 128), 256, 0, stream>>>(
      x_bf, wq_bf, q_f32, TOKENS, QDIM, HIDDEN);
  gemm_bf16_kernel<<<dim3(KVDIM / 128, TOKENS / 128), 256, 0, stream>>>(
      x_bf, wk_bf, k_f32, TOKENS, KVDIM, HIDDEN);
  gemm_bf16_kernel<<<dim3(KVDIM / 128, TOKENS / 128), 256, 0, stream>>>(
      x_bf, wv_bf, v_f32, TOKENS, KVDIM, HIDDEN);

  // RoPE + head-major transpose + bf16.
  {
    size_t nq = (size_t)TOKENS * QDIM;
    rope_transpose_kernel<<<(unsigned)((nq + 255) / 256), 256, 0, stream>>>(
        q_f32, pos, cosT, sinT, q_bf, NHEADS);
    size_t nk = (size_t)TOKENS * KVDIM;
    rope_transpose_kernel<<<(unsigned)((nk + 255) / 256), 256, 0, stream>>>(
        k_f32, pos, cosT, sinT, k_bf, NKVH);
    transpose_bf16_kernel<<<(unsigned)((nk + 255) / 256), 256, 0, stream>>>(
        v_f32, v_bf, NKVH);
  }

  // Flash attention: one wave per (b, head, 16-row q tile).
  flash_attn_kernel<<<BB * NHEADS * (SS / 16), 32, 0, stream>>>(
      q_bf, k_bf, v_bf, attn_bf);

  // Output projection.
  gemm_bf16_kernel<<<dim3(HIDDEN / 128, TOKENS / 128), 256, 0, stream>>>(
      attn_bf, wo_bf, out, TOKENS, HIDDEN, QDIM);
}
